// MultiTaskGAT_48490180772397
// MI455X (gfx1250) — matrix operand order repaired and checked by reading, not compiled, not run on hardware
//
#include <hip/hip_runtime.h>
#include <hip/hip_bf16.h>
#include <cstdint>

typedef __attribute__((ext_vector_type(16))) _Float16 v16h;
typedef __attribute__((ext_vector_type(8)))  _Float16 v8h;
typedef __attribute__((ext_vector_type(8)))  float    v8f;

constexpr int NN    = 20000;
constexpr int FIN   = 512;
constexpr int EE    = 640000;
constexpr int ETOT  = EE + NN;   // with self loops
constexpr int HEADS = 8;
constexpr int HID   = 16;

// ---------------------------------------------------------------- helpers
__device__ __forceinline__ unsigned ford(float f) {
  unsigned u = __float_as_uint(f);
  return (u & 0x80000000u) ? ~u : (u | 0x80000000u);
}
__device__ __forceinline__ float ford_inv(unsigned o) {
  unsigned u = (o & 0x80000000u) ? (o & 0x7FFFFFFFu) : ~o;
  return __uint_as_float(u);
}
__device__ __forceinline__ float lrelu(float x) { return x > 0.f ? x : 0.2f * x; }

// ---------------------------------------------------------------- casts / packing
__global__ void k_cast_half(const float* __restrict__ s, _Float16* __restrict__ d, int n) {
  int i = blockIdx.x * blockDim.x + threadIdx.x;
  if (i < n) d[i] = (_Float16)s[i];
}

// W[K,N] (row major) -> Wt[N,K] f16 (row major)
__global__ void k_transpose_half(const float* __restrict__ W, _Float16* __restrict__ Wt,
                                 int K, int N) {
  int i = blockIdx.x * blockDim.x + threadIdx.x;
  if (i >= N * K) return;
  int n = i / K, k = i % K;
  Wt[i] = (_Float16)W[(size_t)k * N + n];
}

// ---------------------------------------------------------------- WMMA GEMM
// D[M,N] = A[M,K] * Bt[N,K]^T  (f16 in, f32 out).  M%16==0, K%32==0, N%16==0.
// One wave per 16x16 tile; 4 waves/block tiled along N.
__global__ void k_wmma_gemm(const _Float16* __restrict__ A,
                            const _Float16* __restrict__ Bt,
                            float* __restrict__ D,
                            int M, int N, int K) {
  const int lane   = threadIdx.x & 31;
  const int wave   = threadIdx.x >> 5;
  const int n_tile = blockIdx.y * 4 + wave;
  if (n_tile * 16 >= N) return;                  // wave-uniform; EXEC stays all-ones

  const int m0   = blockIdx.x * 16;
  const int mrow = m0 + (lane & 15);
  const int kbA  = (lane >> 4) * 16;             // A: lanes 0-15 K 0..15, lanes 16-31 K 16..31
  const int ncol = n_tile * 16 + (lane & 15);
  const int kbB  = (lane >> 4) * 16;             // B: lanes 0-15 K 0..15, lanes 16-31 K 16..31

  const _Float16* Ap = A  + (size_t)mrow * K + kbA;
  const _Float16* Bp = Bt + (size_t)ncol * K + kbB;

  v8f acc = {};
  #pragma unroll 4
  for (int k0 = 0; k0 < K; k0 += 32) {
    v8h a0 = *(const v8h*)(Ap + k0);             // K = kbA + 0..7
    v8h a1 = *(const v8h*)(Ap + k0 + 8);         // K = kbA + 8..15
    v16h b = *(const v16h*)(Bp + k0);            // K = kbB + 0..15 (contiguous in Wt)
    v16h a;
    #pragma unroll
    for (int i = 0; i < 8; ++i) { a[i] = a0[i]; a[i + 8] = a1[i]; }
    acc = __builtin_amdgcn_wmma_f32_16x16x32_f16(false, a, false, b,
                                                 (short)0, acc, false, false);
  }

  const int rbase = m0 + ((lane >> 4) << 3);     // C/D: lanes 16-31 hold M=8..15
  #pragma unroll
  for (int r = 0; r < 8; ++r)
    D[(size_t)(rbase + r) * N + ncol] = acc[r];
}

// ---------------------------------------------------------------- attention logits
// al_s[n,h] = sum_c xl[n,h,c]*a_src[h,c] ; same for al_d
__global__ void k_compute_al(const float* __restrict__ xl,
                             const float* __restrict__ a_src,
                             const float* __restrict__ a_dst,
                             float* __restrict__ als, float* __restrict__ ald,
                             int heads, int out_ch) {
  int t = blockIdx.x * blockDim.x + threadIdx.x;
  if (t >= NN * heads) return;
  int n = t / heads, h = t % heads;
  int C = heads * out_ch;
  const float* xp = xl + (size_t)n * C + h * out_ch;
  const float* sp = a_src + h * out_ch;
  const float* dp = a_dst + h * out_ch;
  float ss = 0.f, sd = 0.f;
  for (int c = 0; c < out_ch; ++c) { float v = xp[c]; ss += v * sp[c]; sd += v * dp[c]; }
  als[t] = ss; ald[t] = sd;
}

// ---------------------------------------------------------------- edge phase 1: segment max
__global__ void k_edge_max(const long long* __restrict__ ei,
                           const float* __restrict__ als, const float* __restrict__ ald,
                           unsigned* __restrict__ mmax, int heads) {
  int eid = blockIdx.x * blockDim.x + threadIdx.x;
  if (eid >= ETOT) return;
  int s, d;
  if (eid < EE) { s = (int)ei[eid]; d = (int)ei[EE + eid]; }
  else          { s = d = eid - EE; }
  for (int h = 0; h < heads; ++h) {
    float e = lrelu(als[s * heads + h] + ald[d * heads + h]);
    atomicMax(&mmax[d * heads + h], ford(e));
  }
}

__global__ void k_decode_max(const unsigned* __restrict__ u, float* __restrict__ f, int n) {
  int i = blockIdx.x * blockDim.x + threadIdx.x;
  if (i < n) f[i] = ford_inv(u[i]);
}

// ---------------------------------------------------------------- edge phase 2: exp-sum
__global__ void k_edge_expsum(const long long* __restrict__ ei,
                              const float* __restrict__ als, const float* __restrict__ ald,
                              const float* __restrict__ mfl, float* __restrict__ denom,
                              int heads) {
  int eid = blockIdx.x * blockDim.x + threadIdx.x;
  if (eid >= ETOT) return;
  int s, d;
  if (eid < EE) { s = (int)ei[eid]; d = (int)ei[EE + eid]; }
  else          { s = d = eid - EE; }
  for (int h = 0; h < heads; ++h) {
    float e = lrelu(als[s * heads + h] + ald[d * heads + h]);
    atomicAdd(&denom[d * heads + h], __expf(e - mfl[d * heads + h]));
  }
}

// ---------------------------------------------------------------- edge phase 3: weighted scatter
// one thread per (edge, 16-channel group); h = g / (out_ch/16)
__global__ void k_edge_agg(const long long* __restrict__ ei,
                           const float* __restrict__ als, const float* __restrict__ ald,
                           const float* __restrict__ mfl, const float* __restrict__ denom,
                           const float* __restrict__ xl, float* __restrict__ out,
                           int heads, int out_ch, int C, int groups) {
  int t = blockIdx.x * blockDim.x + threadIdx.x;
  int eid = t / groups, g = t % groups;
  if (eid >= ETOT) return;
  int s, d;
  if (eid < EE) { s = (int)ei[eid]; d = (int)ei[EE + eid]; }
  else          { s = d = eid - EE; }
  int h = g / (out_ch >> 4);
  float e = lrelu(als[s * heads + h] + ald[d * heads + h]);
  float alpha = __expf(e - mfl[d * heads + h]) / denom[d * heads + h];
  const float4* xs = (const float4*)(xl + (size_t)s * C + g * 16);
  float* od = out + (size_t)d * C + g * 16;
  #pragma unroll
  for (int q = 0; q < 4; ++q) {
    float4 v = xs[q];
    atomicAdd(od + q * 4 + 0, alpha * v.x);
    atomicAdd(od + q * 4 + 1, alpha * v.y);
    atomicAdd(od + q * 4 + 2, alpha * v.z);
    atomicAdd(od + q * 4 + 3, alpha * v.w);
  }
}

// ---------------------------------------------------------------- epilogues
__global__ void k_bias_elu_cast(const float* __restrict__ a, const float* __restrict__ b,
                                _Float16* __restrict__ o, int total, int C) {
  int i = blockIdx.x * blockDim.x + threadIdx.x;
  if (i >= total) return;
  float v = a[i] + b[i % C];
  v = v > 0.f ? v : (__expf(v) - 1.0f);          // ELU(alpha=1)
  o[i] = (_Float16)v;
}

__global__ void k_bias_add(float* __restrict__ o, const float* __restrict__ b,
                           int total, int C) {
  int i = blockIdx.x * blockDim.x + threadIdx.x;
  if (i < total) o[i] += b[i % C];
}

// ---------------------------------------------------------------- host-side conv driver
static void gat_conv(const _Float16* Ah, int K, const _Float16* Wt,
                     const float* a_src, const float* a_dst,
                     int heads, int out_ch, const long long* ei,
                     float* xl, float* als, float* ald,
                     unsigned* mmax, float* denom, float* out_agg,
                     hipStream_t stream) {
  const int C = heads * out_ch;
  dim3 gg((NN + 15) / 16, (C + 63) / 64);
  k_wmma_gemm<<<gg, 128, 0, stream>>>(Ah, Wt, xl, NN, C, K);

  const int nh = NN * heads;
  k_compute_al<<<(nh + 255) / 256, 256, 0, stream>>>(xl, a_src, a_dst, als, ald, heads, out_ch);

  hipMemsetAsync(mmax, 0, (size_t)nh * 4, stream);   // ford() min is 0 -> always replaced
  hipMemsetAsync(denom, 0, (size_t)nh * 4, stream);

  k_edge_max<<<(ETOT + 255) / 256, 256, 0, stream>>>(ei, als, ald, mmax, heads);
  k_decode_max<<<(nh + 255) / 256, 256, 0, stream>>>(mmax, (float*)mmax, nh);
  k_edge_expsum<<<(ETOT + 255) / 256, 256, 0, stream>>>(ei, als, ald, (const float*)mmax,
                                                        denom, heads);
  hipMemsetAsync(out_agg, 0, (size_t)NN * C * 4, stream);
  const int groups = C / 16;
  const long long total = (long long)ETOT * groups;
  k_edge_agg<<<(unsigned)((total + 255) / 256), 256, 0, stream>>>(
      ei, als, ald, (const float*)mmax, denom, xl, out_agg, heads, out_ch, C, groups);
}

// ---------------------------------------------------------------- entry
extern "C" void kernel_launch(void* const* d_in, const int* in_sizes, int n_in,
                              void* d_out, int out_size, void* d_ws, size_t ws_size,
                              hipStream_t stream) {
  (void)in_sizes; (void)n_in; (void)out_size; (void)ws_size;

  const float*     x    = (const float*)d_in[0];
  const long long* ei   = (const long long*)d_in[1];   // int64 [2,E]
  const float* W1  = (const float*)d_in[2];
  const float* as1 = (const float*)d_in[3];
  const float* ad1 = (const float*)d_in[4];
  const float* b1  = (const float*)d_in[5];
  const float* W2  = (const float*)d_in[6];
  const float* as2 = (const float*)d_in[7];
  const float* ad2 = (const float*)d_in[8];
  const float* b2  = (const float*)d_in[9];
  const float* Wm  = (const float*)d_in[10];
  const float* asm_= (const float*)d_in[11];
  const float* adm = (const float*)d_in[12];
  const float* bm  = (const float*)d_in[13];
  const float* Wi  = (const float*)d_in[14];
  const float* asi = (const float*)d_in[15];
  const float* adi = (const float*)d_in[16];
  const float* bi  = (const float*)d_in[17];
  const float* Wc  = (const float*)d_in[18];
  const float* asc = (const float*)d_in[19];
  const float* adc = (const float*)d_in[20];
  const float* bc  = (const float*)d_in[21];
  float* out = (float*)d_out;

  // workspace carve-up (256B aligned)
  size_t off = 0; char* wsb = (char*)d_ws;
  auto alloc = [&](size_t bytes) -> void* {
    void* p = wsb + off; off += (bytes + 255) & ~(size_t)255; return p;
  };
  _Float16* xh  = (_Float16*)alloc((size_t)NN * FIN * 2);   // x in f16
  _Float16* hh  = (_Float16*)alloc((size_t)NN * 128 * 2);   // h1 then h2 in f16
  _Float16* W1t = (_Float16*)alloc((size_t)128 * 512 * 2);
  _Float16* W2t = (_Float16*)alloc((size_t)128 * 128 * 2);
  _Float16* Wmt = (_Float16*)alloc((size_t)128 * 128 * 2);
  _Float16* Wit = (_Float16*)alloc((size_t)64  * 128 * 2);
  _Float16* Wct = (_Float16*)alloc((size_t)32  * 128 * 2);
  float*    xl  = (float*)alloc((size_t)NN * 128 * 4);
  float*    als = (float*)alloc((size_t)NN * HEADS * 4);
  float*    ald = (float*)alloc((size_t)NN * HEADS * 4);
  unsigned* mmx = (unsigned*)alloc((size_t)NN * HEADS * 4);
  float*    den = (float*)alloc((size_t)NN * HEADS * 4);
  float*    agg = (float*)alloc((size_t)NN * 128 * 4);

  // pack f16 operands
  {
    int n = NN * FIN;
    k_cast_half<<<(n + 255) / 256, 256, 0, stream>>>(x, xh, n);
    k_transpose_half<<<(128 * 512 + 255) / 256, 256, 0, stream>>>(W1, W1t, 512, 128);
    k_transpose_half<<<(128 * 128 + 255) / 256, 256, 0, stream>>>(W2, W2t, 128, 128);
    k_transpose_half<<<(128 * 128 + 255) / 256, 256, 0, stream>>>(Wm, Wmt, 128, 128);
    k_transpose_half<<<(128 * 64  + 255) / 256, 256, 0, stream>>>(Wi, Wit, 128, 64);
    k_transpose_half<<<(128 * 32  + 255) / 256, 256, 0, stream>>>(Wc, Wct, 128, 32);
  }

  // ----- layer 1: GAT(512 -> 8x16), ELU
  gat_conv(xh, FIN, W1t, as1, ad1, HEADS, HID, ei, xl, als, ald, mmx, den, agg, stream);
  k_bias_elu_cast<<<(NN * 128 + 255) / 256, 256, 0, stream>>>(agg, b1, hh, NN * 128, 128);

  // ----- layer 2: GAT(128 -> 8x16), ELU
  gat_conv(hh, 128, W2t, as2, ad2, HEADS, HID, ei, xl, als, ald, mmx, den, agg, stream);
  k_bias_elu_cast<<<(NN * 128 + 255) / 256, 256, 0, stream>>>(agg, b2, hh, NN * 128, 128);

  // ----- task heads (aggregate straight into d_out segments)
  float* med  = out;
  float* immp = out + (size_t)NN * 128;
  float* care = immp + (size_t)NN * 64;

  gat_conv(hh, 128, Wmt, asm_, adm, 1, 128, ei, xl, als, ald, mmx, den, med, stream);
  k_bias_add<<<(NN * 128 + 255) / 256, 256, 0, stream>>>(med, bm, NN * 128, 128);

  gat_conv(hh, 128, Wit, asi, adi, 1, 64, ei, xl, als, ald, mmx, den, immp, stream);
  k_bias_add<<<(NN * 64 + 255) / 256, 256, 0, stream>>>(immp, bi, NN * 64, 64);

  gat_conv(hh, 128, Wct, asc, adc, 1, 32, ei, xl, als, ald, mmx, den, care, stream);
  k_bias_add<<<(NN * 32 + 255) / 256, 256, 0, stream>>>(care, bc, NN * 32, 32);
}
